// Decoder_6803228197529
// MI455X (gfx1250) — compile-verified, matrix operand order checked by reading, not run on hardware
//
#include <hip/hip_runtime.h>
#include <hip/hip_bf16.h>
#include <math.h>

typedef __attribute__((ext_vector_type(16))) _Float16 v16h;
typedef __attribute__((ext_vector_type(8)))  _Float16 h8;
typedef __attribute__((ext_vector_type(4)))  _Float16 h4;
typedef __attribute__((ext_vector_type(2)))  _Float16 h2;
typedef __attribute__((ext_vector_type(8)))  float    v8f;

#define GN_FLAG   1
#define RELU_FLAG 2
#define RES_FLAG  4

// ---------------------------------------------------------------------------
// f32 -> f16 bulk convert (f0 once per launch)
// ---------------------------------------------------------------------------
__global__ __launch_bounds__(256) void f32_to_f16_k(
    const float* __restrict__ src, _Float16* __restrict__ dst, int n)
{
  const int i = blockIdx.x * 256 + threadIdx.x;
  if (i < n) dst[i] = (_Float16)src[i];
}

// ---------------------------------------------------------------------------
// Pack f32 weight (O x I) into zero-padded f16 (Opad x Ipad), row stride Ipad.
// ---------------------------------------------------------------------------
__global__ __launch_bounds__(256) void pack_w_k(
    const float* __restrict__ src, _Float16* __restrict__ dst,
    int O, int I, int Ipad, int total)
{
  const int idx = blockIdx.x * 256 + threadIdx.x;
  if (idx >= total) return;
  const int o = idx / Ipad, k = idx - o * Ipad;
  dst[idx] = (o < O && k < I) ? (_Float16)src[(size_t)o * I + k] : (_Float16)0.f;
}

// ---------------------------------------------------------------------------
// WMMA GEMM (all-f16, tile-aligned):  Y[b,o,n] = epi( sum_k W[o,k]*X[b,k,n] )
// W: packed Opad x Ipad (zero-filled).  X: B x Ipad x N (pad rows zero).
// 64(O) x 64(N) tile per 256-thread workgroup (8 wave32 waves).
// Wave w: N-subtile wn = w>>1, two O-subtiles (w&1)*2 and (w&1)*2+1.
// Loop body is branch-free: b128/b64 copies -> LDS, ds_load_b128 fragments,
// 2x v_wmma_f32_16x16x32_f16. Only epilogue store is masked by o<O.
// Requires: Ipad % 32 == 0, N % 64 == 0, gridDim.y == Opad/64.
// ---------------------------------------------------------------------------
__global__ __launch_bounds__(256) void gemm_mlp_wmma(
    const _Float16* __restrict__ W, const _Float16* __restrict__ X,
    const float* __restrict__ gamma, const float* __restrict__ beta,
    const _Float16* __restrict__ res, _Float16* __restrict__ Y,
    int O, int Ipad, int N, int flags)
{
  constexpr int LD = 40;                  // half-element stride (16B-aligned rows)
  __shared__ _Float16 As[64 * LD];        // As[m][k], k contiguous
  __shared__ _Float16 Bs[64 * LD];        // Bs[n][k], k contiguous

  const int b    = blockIdx.z;
  const int o0   = blockIdx.y * 64;
  const int n0   = blockIdx.x * 64;
  const int t    = threadIdx.x;
  const int lane = t & 31, wave = t >> 5;
  const int wn   = wave >> 1;             // 0..3 : N subtile
  const int wm0  = (wave & 1) * 2;        // O subtiles wm0, wm0+1
  const int half_ = lane >> 4;
  const int l16   = lane & 15;

  // staging coordinates (fixed per thread)
  const int am = t >> 2,  akc = (t & 3) * 8;   // A: row, k-chunk of 8
  const int bnq = (t & 15) * 4;                // B: n-quad base
  const int bkp = (t >> 4) * 2;                // B: k-pair base

  const _Float16* __restrict__ Xb = X + (size_t)b * Ipad * N;
  const _Float16* __restrict__ Wr = W + (size_t)(o0 + am) * Ipad;

  v8f acc0 = {}, acc1 = {};
  const int ksteps = Ipad / 32;
  for (int ks = 0; ks < ksteps; ++ks) {
    const int kk = ks * 32;
    // ---- stage A: 64 x 32 tile of W (pure aligned copy) ----
    *(h8*)(&As[am * LD + akc]) = *(const h8*)(Wr + kk + akc);
    // ---- stage B: 32 x 64 tile of X, transposed into Bs[n][k] ----
    {
      const int k0 = kk + bkp;
      const h4 x0 = *(const h4*)(Xb + (size_t)k0 * N + n0 + bnq);
      const h4 x1 = *(const h4*)(Xb + (size_t)(k0 + 1) * N + n0 + bnq);
      #pragma unroll
      for (int j = 0; j < 4; ++j) {
        h2 pr; pr[0] = x0[j]; pr[1] = x1[j];
        *(h2*)(&Bs[(bnq + j) * LD + bkp]) = pr;
      }
    }
    if (ks + 1 < ksteps) {  // gfx1250 global_prefetch_b8 of next K tile
      const _Float16* pf = Xb + (size_t)(kk + 32) * N + n0 + (t & 63);
      __builtin_prefetch(pf, 0, 1);
    }
    __syncthreads();

    // Fragment loads per ISA 16-bit 16x32 layout:
    // lane (half=lane>>4, r=lane&15): elements 0..7 <- K = half*8+e,
    //                                 elements 8..15 <- K = 16+half*8+e.
    v16h afrag0, afrag1, bfrag;
    {
      const _Float16* bp = &Bs[(wn * 16 + l16) * LD + half_ * 8];
      #pragma unroll
      for (int e = 0; e < 8; ++e) { bfrag[e] = bp[e]; bfrag[8 + e] = bp[16 + e]; }
      const _Float16* a0 = &As[((wm0    ) * 16 + l16) * LD + half_ * 8];
      const _Float16* a1 = &As[((wm0 + 1) * 16 + l16) * LD + half_ * 8];
      #pragma unroll
      for (int e = 0; e < 8; ++e) {
        afrag0[e] = a0[e]; afrag0[8 + e] = a0[16 + e];
        afrag1[e] = a1[e]; afrag1[8 + e] = a1[16 + e];
      }
    }
    acc0 = __builtin_amdgcn_wmma_f32_16x16x32_f16(false, afrag0, false, bfrag,
                                                  (short)0, acc0, false, false);
    acc1 = __builtin_amdgcn_wmma_f32_16x16x32_f16(false, afrag1, false, bfrag,
                                                  (short)0, acc1, false, false);
    __syncthreads();
  }

  // epilogue + store (C/D layout: m = r + half*8, n = lane&15)
  const int n = n0 + wn * 16 + l16;
  _Float16* __restrict__ Yb = Y + (size_t)b * O * N;
  const _Float16* __restrict__ Rb = res ? res + (size_t)b * O * N : nullptr;
  #pragma unroll
  for (int r = 0; r < 8; ++r) {
    const int oA = o0 + (wm0    ) * 16 + half_ * 8 + r;
    const int oB = o0 + (wm0 + 1) * 16 + half_ * 8 + r;
    if (oA < O) {
      float v = acc0[r];
      if (flags & GN_FLAG)   v = gamma[oA] * v + beta[oA];
      if (flags & RES_FLAG)  v += (float)Rb[(size_t)oA * N + n];
      if (flags & RELU_FLAG) v = fmaxf(v, 0.f);
      Yb[(size_t)oA * N + n] = (_Float16)v;
    }
    if (oB < O) {
      float v = acc1[r];
      if (flags & GN_FLAG)   v = gamma[oB] * v + beta[oB];
      if (flags & RES_FLAG)  v += (float)Rb[(size_t)oB * N + n];
      if (flags & RELU_FLAG) v = fmaxf(v, 0.f);
      Yb[(size_t)oB * N + n] = (_Float16)v;
    }
  }
}

// ---------------------------------------------------------------------------
// 3-NN inverse-distance interpolation + channel concat (f16 out).
// Writes Cpad rows: [0,C1)=p1, [C1,C1+C2)=interp(p2), [C1+C2,Cpad)=zero pad.
// ---------------------------------------------------------------------------
__global__ __launch_bounds__(256) void interp_concat(
    const float* __restrict__ xyz1, const float* __restrict__ xyz2,
    const float* __restrict__ p1, const _Float16* __restrict__ p2,
    _Float16* __restrict__ Xc, int N, int S, int C1, int C2, int Cpad)
{
  __shared__ float sx[1024 * 3];          // S <= 1024
  const int b = blockIdx.y;
  for (int i = threadIdx.x; i < S * 3; i += 256)
    sx[i] = xyz2[(size_t)b * S * 3 + i];
  __syncthreads();

  const int n = blockIdx.x * 256 + threadIdx.x;
  if (n >= N) return;
  const float px = xyz1[((size_t)b * N + n) * 3 + 0];
  const float py = xyz1[((size_t)b * N + n) * 3 + 1];
  const float pz = xyz1[((size_t)b * N + n) * 3 + 2];

  float d0 = 1e30f, d1 = 1e30f, d2 = 1e30f;
  int   i0 = 0, i1 = 0, i2 = 0;
  for (int s = 0; s < S; ++s) {
    const float dx = sx[s * 3] - px, dy = sx[s * 3 + 1] - py, dz = sx[s * 3 + 2] - pz;
    const float d = dx * dx + dy * dy + dz * dz;
    if (d < d0)      { d2 = d1; i2 = i1; d1 = d0; i1 = i0; d0 = d; i0 = s; }
    else if (d < d1) { d2 = d1; i2 = i1; d1 = d;  i1 = s; }
    else if (d < d2) { d2 = d;  i2 = s; }
  }
  float w0 = 1.f / (d0 + 1e-8f), w1 = 1.f / (d1 + 1e-8f), w2 = 1.f / (d2 + 1e-8f);
  const float ws = w0 + w1 + w2; w0 /= ws; w1 /= ws; w2 /= ws;

  const float*    __restrict__ p1b = p1 + (size_t)b * C1 * N;
  const _Float16* __restrict__ p2b = p2 + (size_t)b * C2 * S;
  _Float16* __restrict__ Xb = Xc + (size_t)b * Cpad * N;
  for (int c = 0; c < C1; ++c) Xb[(size_t)c * N + n] = (_Float16)p1b[(size_t)c * N + n];
  for (int c = 0; c < C2; ++c) {
    const _Float16* r = p2b + (size_t)c * S;
    Xb[(size_t)(C1 + c) * N + n] =
        (_Float16)(w0 * (float)r[i0] + w1 * (float)r[i1] + w2 * (float)r[i2]);
  }
  for (int c = C1 + C2; c < Cpad; ++c) Xb[(size_t)c * N + n] = (_Float16)0.f;
}

// max_s relu(g * (W @ f)[o,s] + b)  -> outcat[b, lvl*64 + o]   (f32 side path)
__global__ __launch_bounds__(64) void gmp_level(
    const float* __restrict__ f, const float* __restrict__ Wg,
    const float* __restrict__ bg, const float* __restrict__ gg,
    float* __restrict__ outcat, int C, int S, int lvl)
{
  const int b = blockIdx.x, o = threadIdx.x;
  const float* fb = f + (size_t)b * C * S;
  const float* wr = Wg + (size_t)o * C;
  float m = -1e30f;
  for (int s = 0; s < S; ++s) {
    float acc = 0.f;
    for (int c = 0; c < C; ++c) acc += wr[c] * fb[(size_t)c * S + s];
    m = fmaxf(m, fmaxf(gg[o] * acc + bg[o], 0.f));
  }
  outcat[(size_t)b * 320 + lvl * 64 + o] = m;
}

__global__ __launch_bounds__(64) void gmp_end_k(
    const float* __restrict__ g, const float* __restrict__ W,
    const float* __restrict__ bb, const float* __restrict__ gg,
    float* __restrict__ ctx)
{
  const int b = blockIdx.x, o = threadIdx.x;
  const float* gb = g + (size_t)b * 320;
  const float* wr = W + (size_t)o * 320;
  float acc = 0.f;
  for (int j = 0; j < 320; ++j) acc += wr[j] * gb[j];
  ctx[b * 64 + o] = fmaxf(gg[o] * acc + bb[o], 0.f);
}

__global__ __launch_bounds__(64) void cls_k(
    const float* __restrict__ lbl,
    const float* __restrict__ W0, const float* __restrict__ b0, const float* __restrict__ g0,
    const float* __restrict__ W1, const float* __restrict__ b1, const float* __restrict__ g1,
    float* __restrict__ ct)
{
  __shared__ float h1[64];
  const int b = blockIdx.x, o = threadIdx.x;
  float acc = 0.f;
  for (int j = 0; j < 16; ++j) acc += W0[o * 16 + j] * lbl[b * 16 + j];
  h1[o] = fmaxf(g0[o] * acc + b0[o], 0.f);
  __syncthreads();
  acc = 0.f;
  for (int j = 0; j < 64; ++j) acc += W1[o * 64 + j] * h1[j];
  ct[b * 64 + o] = fmaxf(g1[o] * acc + b1[o], 0.f);
}

// feat[b, 0:128, n]=x ; [128:192]=ctx bcast ; [192:256]=ct bcast   (N=2048)
__global__ __launch_bounds__(256) void feat_k(
    const _Float16* __restrict__ x, const float* __restrict__ ctx,
    const float* __restrict__ ct, _Float16* __restrict__ feat)
{
  const int b = blockIdx.y;
  const int idx = blockIdx.x * 256 + threadIdx.x;   // 0 .. 256*2048-1
  const int c = idx >> 11, n = idx & 2047;
  _Float16 v;
  if (c < 128)      v = x[(size_t)b * 128 * 2048 + (size_t)c * 2048 + n];
  else if (c < 192) v = (_Float16)ctx[b * 64 + (c - 128)];
  else              v = (_Float16)ct[b * 64 + (c - 192)];
  feat[(size_t)b * 524288 + idx] = v;
}

// out[b, n, k] = log_softmax over k of logits[b, k, n]  (K=50, N=2048)
__global__ __launch_bounds__(256) void logsoftmax_k(
    const _Float16* __restrict__ logits, float* __restrict__ out)
{
  const int b = blockIdx.y;
  const int n = blockIdx.x * 256 + threadIdx.x;
  const _Float16* lb = logits + (size_t)b * 50 * 2048 + n;
  float v[50], mx = -1e30f;
  for (int k = 0; k < 50; ++k) { v[k] = (float)lb[(size_t)k * 2048]; mx = fmaxf(mx, v[k]); }
  float s = 0.f;
  for (int k = 0; k < 50; ++k) s += expf(v[k] - mx);
  const float ls = mx + logf(s);
  float* ob = out + ((size_t)b * 2048 + n) * 50;
  for (int k = 0; k < 50; ++k) ob[k] = v[k] - ls;
}

// ---------------------------------------------------------------------------
extern "C" void kernel_launch(void* const* d_in, const int* in_sizes, int n_in,
                              void* d_out, int out_size, void* d_ws, size_t ws_size,
                              hipStream_t stream) {
  (void)in_sizes; (void)n_in; (void)out_size; (void)ws_size;
  static const int DEc[5] = {128, 512, 256, 128, 128};
  static const int CLc[5] = {128, 128, 64, 32, 16};
  static const int SLc[5] = {128, 256, 512, 1024, 2048};

  const float* xyz[5] = {(const float*)d_in[0], (const float*)d_in[2], (const float*)d_in[4],
                         (const float*)d_in[6], (const float*)d_in[8]};
  const float* fs[5]  = {(const float*)d_in[1], (const float*)d_in[3], (const float*)d_in[5],
                         (const float*)d_in[7], (const float*)d_in[9]};
  const float* cls_label = (const float*)d_in[10];
  const float* P = (const float*)d_in[11];

  // ---- param offsets: flat buffer in JAX pytree order (dict keys sorted) ---
  size_t cur_off = 0;
  auto take = [&](size_t n) { const size_t o = cur_off; cur_off += n; return o; };
  struct MLPo { size_t W, b, g; };
  // 'clf' (keys: W1, W2, b1, g1)
  const size_t o_clfW1 = take(128 * 256);
  const size_t o_clfW2 = take(50 * 128);
  const size_t o_clfb1 = take(128);
  const size_t o_clfg1 = take(128);
  // 'cls' (each mlp: W, b, g)
  MLPo cls0{take(64 * 16), take(64), take(64)};
  MLPo cls1{take(64 * 64), take(64), take(64)};
  // 'fp' : per level {blocks:[{W2,b2,g2,m1:{W,b,g}} x2], fuse:{W,b,g}}
  struct Blk { size_t W2, b2, g2; MLPo m1; };
  struct FPp { Blk blk[2]; MLPo fuse; } fp[4];
  for (int i = 0; i < 4; ++i) {
    const int cin = DEc[i] + CLc[i + 1], cout = DEc[i + 1], hid = cout / 4;
    for (int j = 0; j < 2; ++j) {
      fp[i].blk[j].W2 = take((size_t)cout * hid);
      fp[i].blk[j].b2 = take(cout);
      fp[i].blk[j].g2 = take(cout);
      fp[i].blk[j].m1 = MLPo{take((size_t)hid * cout), take(hid), take(hid)};
    }
    fp[i].fuse = MLPo{take((size_t)cout * cin), take(cout), take(cout)};
  }
  // 'gmp', 'gmp_end'
  MLPo gmpp[5];
  for (int i = 0; i < 5; ++i) gmpp[i] = MLPo{take((size_t)64 * CLc[i]), take(64), take(64)};
  MLPo gend = MLPo{take(64 * 320), take(64), take(64)};

  // ---- workspace carve-up ----
  char* ws = (char*)d_ws;
  size_t off = 0;
  auto allocH = [&](size_t n) {
    void* p = ws + off; off += n * sizeof(_Float16); off = (off + 255) & ~(size_t)255;
    return (_Float16*)p;
  };
  auto allocF = [&](size_t n) {
    void* p = ws + off; off += n * sizeof(float); off = (off + 255) & ~(size_t)255;
    return (float*)p;
  };
  _Float16* wpool  = allocH(720896);              // packed f16 weights
  _Float16* f0h    = allocH((size_t)32 * 16384);  // f16 copy of f0
  _Float16* p0     = allocH((size_t)32 * 262144); // ping
  _Float16* p1     = allocH((size_t)32 * 262144); // pong
  _Float16* hbuf   = allocH((size_t)32 * 65536);  // bottleneck hidden
  _Float16* xcat   = allocH((size_t)32 * 327680); // interp concat (Cpad<=576 rows)
  _Float16* feat   = allocH((size_t)32 * 524288);
  _Float16* logits = allocH((size_t)32 * 102400); // 50 x 2048
  float*    gmpcat = allocF((size_t)32 * 320);
  float*    ctx    = allocF((size_t)32 * 64);
  float*    ct     = allocF((size_t)32 * 64);

  // ---- pack weights (zero-padded Opad x Ipad f16 tiles) ----
  auto rup = [](int x, int m) { return (x + m - 1) / m * m; };
  size_t woff = 0;
  auto pack = [&](size_t srcOff, int O, int I) {
    const int Opad = rup(O, 64), Ipad = rup(I, 32);
    _Float16* dst = wpool + woff;
    const int total = Opad * Ipad;
    woff += (size_t)total;
    pack_w_k<<<(total + 255) / 256, 256, 0, stream>>>(P + srcOff, dst, O, I, Ipad, total);
    return dst;
  };
  const _Float16 *wFuse[4], *wM1[4][2], *wW2[4][2];
  for (int i = 0; i < 4; ++i) {
    const int cin = DEc[i] + CLc[i + 1], cout = DEc[i + 1], hid = cout / 4;
    wFuse[i] = pack(fp[i].fuse.W, cout, cin);
    for (int j = 0; j < 2; ++j) {
      wM1[i][j] = pack(fp[i].blk[j].m1.W, hid, cout);
      wW2[i][j] = pack(fp[i].blk[j].W2, cout, hid);
    }
  }
  const _Float16* wClf1 = pack(o_clfW1, 128, 256);
  const _Float16* wClf2 = pack(o_clfW2, 50, 128);

  f32_to_f16_k<<<2048, 256, 0, stream>>>(fs[0], f0h, 32 * 16384);

  // ---- FP chain ----
  const _Float16* curx = f0h;
  _Float16* ping[2] = {p0, p1};
  int curbuf = -1;
  for (int i = 0; i < 4; ++i) {
    const int N = SLc[i + 1], S = SLc[i], C1 = CLc[i + 1], C2 = DEc[i];
    const int cin = C1 + C2, cout = DEc[i + 1], hid = cout / 4;
    const int cinP = rup(cin, 32);

    dim3 gi((N + 255) / 256, 32);
    interp_concat<<<gi, 256, 0, stream>>>(xyz[i + 1], xyz[i], fs[i + 1], curx,
                                          xcat, N, S, C1, C2, cinP);

    const int nxt = (curbuf == 0) ? 1 : 0;
    dim3 gg(N / 64, cout / 64, 32);
    gemm_mlp_wmma<<<gg, 256, 0, stream>>>(wFuse[i], xcat,
                                          P + fp[i].fuse.g, P + fp[i].fuse.b,
                                          nullptr, ping[nxt], cout, cinP, N,
                                          GN_FLAG | RELU_FLAG);
    curbuf = nxt; curx = ping[curbuf];

    for (int j = 0; j < 2; ++j) {
      dim3 gh(N / 64, rup(hid, 64) / 64, 32);
      gemm_mlp_wmma<<<gh, 256, 0, stream>>>(wM1[i][j], curx,
                                            P + fp[i].blk[j].m1.g, P + fp[i].blk[j].m1.b,
                                            nullptr, hbuf, hid, cout, N,
                                            GN_FLAG | RELU_FLAG);
      const int oth = curbuf ^ 1;
      gemm_mlp_wmma<<<gg, 256, 0, stream>>>(wW2[i][j], hbuf,
                                            P + fp[i].blk[j].g2, P + fp[i].blk[j].b2,
                                            curx, ping[oth], cout, rup(hid, 32), N,
                                            GN_FLAG | RES_FLAG | RELU_FLAG);
      curbuf = oth; curx = ping[curbuf];
    }
  }

  // ---- global context + class token (small f32 side paths) ----
  for (int i = 0; i < 5; ++i)
    gmp_level<<<32, 64, 0, stream>>>(fs[i], P + gmpp[i].W, P + gmpp[i].b, P + gmpp[i].g,
                                     gmpcat, CLc[i], SLc[i], i);
  gmp_end_k<<<32, 64, 0, stream>>>(gmpcat, P + gend.W, P + gend.b, P + gend.g, ctx);
  cls_k<<<32, 64, 0, stream>>>(cls_label, P + cls0.W, P + cls0.b, P + cls0.g,
                               P + cls1.W, P + cls1.b, P + cls1.g, ct);

  // ---- classifier ----
  feat_k<<<dim3(2048, 32), 256, 0, stream>>>(curx, ctx, ct, feat);
  _Float16* hclf = xcat;   // reuse (327680 >= 128*2048 per batch)
  gemm_mlp_wmma<<<dim3(32, 2, 32), 256, 0, stream>>>(wClf1, feat,
                                                     P + o_clfg1, P + o_clfb1,
                                                     nullptr, hclf, 128, 256, 2048,
                                                     GN_FLAG);
  gemm_mlp_wmma<<<dim3(32, 1, 32), 256, 0, stream>>>(wClf2, hclf, nullptr, nullptr,
                                                     nullptr, logits, 50, 128, 2048, 0);
  logsoftmax_k<<<dim3(8, 32), 256, 0, stream>>>(logits, (float*)d_out);
}
